// Multi_Granular_Attention_Orchestrator_64252710748730
// MI455X (gfx1250) — matrix operand order repaired and checked by reading, not compiled, not run on hardware
//
#include <hip/hip_runtime.h>
#include <hip/hip_bf16.h>
#include <math.h>

typedef __attribute__((ext_vector_type(16))) _Float16 v16h;
typedef __attribute__((ext_vector_type(8)))  float    v8f;

#define BATCH   2
#define CH      64
#define DD      16
#define HH      64
#define WW      64
#define NTOK    (DD*HH*WW)      // 65536
#define QKV_CH  192
#define HEADS   8

// ---------------------------------------------------------------------------
// WMMA fragment builders for v_wmma_f32_16x16x32_f16 (wave32, CDNA5 layouts)
// A (16xK f16): lane%16 = M row; lanes 0-15 hold K = {k0..k0+7, k0+16..k0+23},
//               lanes 16-31 hold K = {k0+8..k0+15, k0+24..k0+31}
// B (Kx16 f16): lane%16 = N col; lanes 0-15 hold K = {k0..k0+7, k0+16..k0+23},
//               lanes 16-31 hold K = {k0+8..k0+15, k0+24..k0+31}
// D (16x16 f32, 8 VGPRs): elem r -> M = r + 8*(lane>=16), N = lane%16
// ---------------------------------------------------------------------------
static __device__ __forceinline__ v16h frag_a(const _Float16* Ws, int ld,
                                              int m0, int k0, int lane) {
  v16h a;
  const _Float16* row = Ws + (size_t)(m0 + (lane & 15)) * ld + k0 + (((lane >> 4) & 1) << 3);
  #pragma unroll
  for (int e = 0; e < 16; ++e)
    a[e] = row[((e & 8) << 1) + (e & 7)];
  return a;
}

static __device__ __forceinline__ v16h frag_b(const _Float16* Bs, int ld,
                                              int k0, int n0, int lane) {
  v16h b;
  const _Float16* p = Bs + (size_t)(k0 + (((lane >> 4) & 1) << 3)) * ld + n0 + (lane & 15);
  #pragma unroll
  for (int e = 0; e < 16; ++e)
    b[e] = p[(size_t)(((e & 8) << 1) + (e & 7)) * ld];
  return b;
}

// ---------------------------------------------------------------------------
// K1: pointwise qkv conv  ->  qkv_h[b][192][n] (f16), via WMMA GEMM
// block = 256 thr (8 waves), covers 64 tokens; 12 Mtiles x 4 Ntiles = 48 tiles
// ---------------------------------------------------------------------------
__global__ __launch_bounds__(256) void k_qkv(const float* __restrict__ x,
                                             const float* __restrict__ w_qkv,
                                             const float* __restrict__ b_qkv,
                                             _Float16* __restrict__ qkv_h) {
  __shared__ _Float16 Wq[QKV_CH][72];   // padded A (conflict-free)
  __shared__ _Float16 Bs[64][64];       // x tile, [K][N]
  const int tid = threadIdx.x;
  const int b  = blockIdx.x / (NTOK / 64);
  const int n0 = (blockIdx.x % (NTOK / 64)) * 64;

  for (int i = tid; i < QKV_CH * 64; i += 256)
    Wq[i >> 6][i & 63] = (_Float16)w_qkv[i];
  for (int i = tid; i < 64 * 64; i += 256) {
    int k = i >> 6, c = i & 63;
    Bs[k][c] = (_Float16)x[((size_t)b * CH + k) * NTOK + n0 + c];
  }
  __syncthreads();

  const int wv = tid >> 5, lane = tid & 31;
  for (int t = wv; t < 48; t += 8) {
    const int mt = t >> 2, nt = t & 3;
    v8f c = {};
    #pragma unroll
    for (int k0 = 0; k0 < 64; k0 += 32) {
      v16h a  = frag_a(&Wq[0][0], 72, mt * 16, k0, lane);
      v16h bb = frag_b(&Bs[0][0], 64, k0, nt * 16, lane);
      c = __builtin_amdgcn_wmma_f32_16x16x32_f16(false, a, false, bb,
                                                 (short)0, c, false, false);
    }
    const int col = n0 + nt * 16 + (lane & 15);
    #pragma unroll
    for (int r = 0; r < 8; ++r) {
      const int row = mt * 16 + r + ((lane >> 4) << 3);
      qkv_h[((size_t)b * QKV_CH + row) * NTOK + col] = (_Float16)(c[r] + b_qkv[row]);
    }
  }
}

// ---------------------------------------------------------------------------
// K2: depthwise 3x3x3 conv (groups=192, pad=1) + bias -> qkvdw (fp32)
// ---------------------------------------------------------------------------
__global__ __launch_bounds__(256) void k_dwconv(const _Float16* __restrict__ qkv_h,
                                                const float* __restrict__ w_dw,
                                                const float* __restrict__ b_dw,
                                                float* __restrict__ qkvdw) {
  const long total = (long)BATCH * QKV_CH * NTOK;
  for (long t = (long)blockIdx.x * 256 + threadIdx.x; t < total;
       t += (long)gridDim.x * 256) {
    const int xx = (int)(t & 63);
    const int y  = (int)((t >> 6) & 63);
    const int z  = (int)((t >> 12) & 15);
    const int ch = (int)((t >> 16) % QKV_CH);
    const int b  = (int)(t / ((long)QKV_CH * NTOK));
    const float* wd = w_dw + ch * 27;
    const _Float16* src = qkv_h + ((size_t)b * QKV_CH + ch) * NTOK;
    float acc = b_dw[ch];
    #pragma unroll
    for (int dz = -1; dz <= 1; ++dz) {
      const int zz = z + dz; if (zz < 0 || zz >= DD) continue;
      #pragma unroll
      for (int dy = -1; dy <= 1; ++dy) {
        const int yy = y + dy; if (yy < 0 || yy >= HH) continue;
        #pragma unroll
        for (int dx = -1; dx <= 1; ++dx) {
          const int xc = xx + dx; if (xc < 0 || xc >= WW) continue;
          acc += wd[(dz + 1) * 9 + (dy + 1) * 3 + (dx + 1)] *
                 (float)src[((size_t)zz * HH + yy) * WW + xc];
        }
      }
    }
    qkvdw[t] = acc;
  }
}

// ---------------------------------------------------------------------------
// K3: per (b,head) Gram matrix G=q k^T (8x8) + sum-of-squares norms, reduced
// over all 65536 tokens. stats layout per bh: [0:64)=G, [64:72)=|q|^2, [72:80)=|k|^2
// ---------------------------------------------------------------------------
__global__ __launch_bounds__(64) void k_gram(const float* __restrict__ qkvdw,
                                             float* __restrict__ stats) {
  __shared__ float red[80][64];
  const int chunk = blockIdx.x & 63;
  const int bh = blockIdx.x >> 6;
  const int b = bh >> 3, h = bh & 7;
  const int t = threadIdx.x;
  float G[8][8] = {}; float nq[8] = {}; float nk[8] = {};
  const float* qb = qkvdw + ((size_t)b * QKV_CH + h * 8) * NTOK;
  const float* kb = qb + (size_t)64 * NTOK;
  const int nend = (chunk + 1) * (NTOK / 64);
  for (int n = chunk * (NTOK / 64) + t; n < nend; n += 64) {
    float q[8], kk[8];
    #pragma unroll
    for (int i = 0; i < 8; ++i) {
      q[i]  = qb[(size_t)i * NTOK + n];
      kk[i] = kb[(size_t)i * NTOK + n];
    }
    #pragma unroll
    for (int i = 0; i < 8; ++i) {
      nq[i] += q[i] * q[i];
      nk[i] += kk[i] * kk[i];
      #pragma unroll
      for (int j = 0; j < 8; ++j) G[i][j] += q[i] * kk[j];
    }
  }
  #pragma unroll
  for (int i = 0; i < 8; ++i) {
    #pragma unroll
    for (int j = 0; j < 8; ++j) red[i * 8 + j][t] = G[i][j];
    red[64 + i][t] = nq[i];
    red[72 + i][t] = nk[i];
  }
  __syncthreads();
  for (int v = t; v < 80; v += 64) {
    float s = 0.f;
    for (int l = 0; l < 64; ++l) s += red[v][l];
    atomicAdd(&stats[bh * 80 + v], s);
  }
}

// ---------------------------------------------------------------------------
// K4: normalize Gram -> attn (8x8), temperature, 4x top-k masked softmax,
// merge with softmax(attn_weights) into one combined matrix P[b][h][8][8].
// One thread per (b,h,row). Since out is linear in probs this is exact.
// ---------------------------------------------------------------------------
__global__ __launch_bounds__(128) void k_combine(const float* __restrict__ stats,
                                                 const float* __restrict__ temperature,
                                                 const float* __restrict__ attn_w,
                                                 float* __restrict__ P) {
  const int t = threadIdx.x;
  const int b = t >> 6, h = (t >> 3) & 7, i = t & 7;
  const int bh = b * 8 + h;
  const float* s = stats + bh * 80;
  const float qn = fmaxf(sqrtf(s[64 + i]), 1e-12f);
  const float temp = temperature[h];
  float a[8];
  #pragma unroll
  for (int j = 0; j < 8; ++j) {
    const float kn = fmaxf(sqrtf(s[72 + j]), 1e-12f);
    a[j] = s[i * 8 + j] / (qn * kn) * temp;
  }
  float srt[8];
  #pragma unroll
  for (int j = 0; j < 8; ++j) srt[j] = a[j];
  for (int jj = 1; jj < 8; ++jj) {            // insertion sort, descending
    float key = srt[jj]; int l = jj - 1;
    while (l >= 0 && srt[l] < key) { srt[l + 1] = srt[l]; --l; }
    srt[l + 1] = key;
  }
  const float aw0 = attn_w[0], aw1 = attn_w[1], aw2 = attn_w[2], aw3 = attn_w[3];
  const float m = fmaxf(fmaxf(aw0, aw1), fmaxf(aw2, aw3));
  const float e0 = expf(aw0 - m), e1 = expf(aw1 - m), e2 = expf(aw2 - m), e3 = expf(aw3 - m);
  const float inv = 1.0f / (e0 + e1 + e2 + e3);
  const float sw[4] = { e0 * inv, e1 * inv, e2 * inv, e3 * inv };
  const int kvs[4] = { 4, 5, 6, 6 };          // int(8 * {0.5,0.67,0.75,0.8})
  float outp[8] = {};
  const float mx = srt[0];
  #pragma unroll
  for (int r = 0; r < 4; ++r) {
    const float thr = srt[kvs[r] - 1];
    float den = 0.f;
    #pragma unroll
    for (int j = 0; j < 8; ++j) if (a[j] >= thr) den += expf(a[j] - mx);
    const float rs = sw[r] / den;
    #pragma unroll
    for (int j = 0; j < 8; ++j) if (a[j] >= thr) outp[j] += rs * expf(a[j] - mx);
  }
  #pragma unroll
  for (int j = 0; j < 8; ++j) P[(bh * 8 + i) * 8 + j] = outp[j];
}

// ---------------------------------------------------------------------------
// K5: fused tail. Per wave: 16 tokens, 4 chained WMMA GEMM stages staged
// through a private 128x16 f16 LDS panel:
//   w = Pdense @ v ; r = sigmoid(bn1(Wr@w + br)) ; e = w*r ;
//   t = bn2(Wf @ [e;x] + bf) ; out = Wp @ t + bp
// ---------------------------------------------------------------------------
__global__ __launch_bounds__(128) void k_tail(const float* __restrict__ qkvdw,
    const float* __restrict__ x, const float* __restrict__ P,
    const float* __restrict__ w_recal, const float* __restrict__ b_recal,
    const float* __restrict__ bn1_g, const float* __restrict__ bn1_b,
    const float* __restrict__ w_fus, const float* __restrict__ b_fus,
    const float* __restrict__ bn2_g, const float* __restrict__ bn2_b,
    const float* __restrict__ w_proj, const float* __restrict__ b_proj,
    float* __restrict__ out) {
  __shared__ _Float16 Pd[64][64];
  __shared__ _Float16 Wr[64][64];
  __shared__ _Float16 Wf[64][128];
  __shared__ _Float16 Wp[64][64];
  __shared__ _Float16 act[4][128 * 16];
  const int tid = threadIdx.x;
  const int b = blockIdx.x >> 10;                // NTOK/64 = 1024 blocks/batch
  const int tile0 = (blockIdx.x & 1023) << 6;

  for (int i = tid; i < 64 * 64; i += 128) {
    const int co = i >> 6, ci = i & 63;
    const float pv = ((co >> 3) == (ci >> 3))
        ? P[((b * 8 + (co >> 3)) * 8 + (co & 7)) * 8 + (ci & 7)] : 0.f;
    Pd[co][ci] = (_Float16)pv;
    Wr[co][ci] = (_Float16)w_recal[i];
    Wp[co][ci] = (_Float16)w_proj[i];
  }
  for (int i = tid; i < 64 * 128; i += 128)
    Wf[i >> 7][i & 127] = (_Float16)w_fus[i];
  __syncthreads();

  const int wv = tid >> 5, lane = tid & 31;
  const int p0 = tile0 + wv * 16;
  const int col = lane & 15;
  _Float16* A = &act[wv][0];
  const float inv_std = rsqrtf(1.0f + 1e-5f);

  // stage v (channels 128..191) as B operand [K=64][16]
  for (int i = lane; i < 64 * 16; i += 32)
    A[i] = (_Float16)qkvdw[((size_t)b * QKV_CH + 128 + (i >> 4)) * NTOK + p0 + (i & 15)];
  __syncthreads();

  // GEMM1: weighted = Pdense @ v
  v8f cw[4];
  #pragma unroll
  for (int mt = 0; mt < 4; ++mt) {
    v8f c = {};
    #pragma unroll
    for (int k0 = 0; k0 < 64; k0 += 32) {
      v16h a  = frag_a(&Pd[0][0], 64, mt * 16, k0, lane);
      v16h bb = frag_b(A, 16, k0, 0, lane);
      c = __builtin_amdgcn_wmma_f32_16x16x32_f16(false, a, false, bb,
                                                 (short)0, c, false, false);
    }
    cw[mt] = c;
  }
  __syncthreads();
  #pragma unroll
  for (int mt = 0; mt < 4; ++mt)
    #pragma unroll
    for (int r = 0; r < 8; ++r)
      A[(mt * 16 + r + ((lane >> 4) << 3)) * 16 + col] = (_Float16)cw[mt][r];
  __syncthreads();

  // GEMM2: recal = sigmoid(bn1(Wr @ w + b_recal)); enhanced = w * recal
  #pragma unroll
  for (int mt = 0; mt < 4; ++mt) {
    v8f c = {};
    #pragma unroll
    for (int k0 = 0; k0 < 64; k0 += 32) {
      v16h a  = frag_a(&Wr[0][0], 64, mt * 16, k0, lane);
      v16h bb = frag_b(A, 16, k0, 0, lane);
      c = __builtin_amdgcn_wmma_f32_16x16x32_f16(false, a, false, bb,
                                                 (short)0, c, false, false);
    }
    #pragma unroll
    for (int r = 0; r < 8; ++r) {
      const int row = mt * 16 + r + ((lane >> 4) << 3);
      const float y = (c[r] + b_recal[row]) * (bn1_g[row] * inv_std) + bn1_b[row];
      cw[mt][r] *= 1.0f / (1.0f + expf(-y));
    }
  }
  __syncthreads();
  #pragma unroll
  for (int mt = 0; mt < 4; ++mt)
    #pragma unroll
    for (int r = 0; r < 8; ++r)
      A[(mt * 16 + r + ((lane >> 4) << 3)) * 16 + col] = (_Float16)cw[mt][r];
  for (int i = lane; i < 64 * 16; i += 32)      // residual x -> rows 64..127
    A[(64 + (i >> 4)) * 16 + (i & 15)] =
        (_Float16)x[((size_t)b * CH + (i >> 4)) * NTOK + p0 + (i & 15)];
  __syncthreads();

  // GEMM3 (K=128): t = bn2(Wf @ [e;x] + b_fus)
  v8f ct[4];
  #pragma unroll
  for (int mt = 0; mt < 4; ++mt) {
    v8f c = {};
    #pragma unroll
    for (int k0 = 0; k0 < 128; k0 += 32) {
      v16h a  = frag_a(&Wf[0][0], 128, mt * 16, k0, lane);
      v16h bb = frag_b(A, 16, k0, 0, lane);
      c = __builtin_amdgcn_wmma_f32_16x16x32_f16(false, a, false, bb,
                                                 (short)0, c, false, false);
    }
    #pragma unroll
    for (int r = 0; r < 8; ++r) {
      const int row = mt * 16 + r + ((lane >> 4) << 3);
      c[r] = (c[r] + b_fus[row]) * (bn2_g[row] * inv_std) + bn2_b[row];
    }
    ct[mt] = c;
  }
  __syncthreads();
  #pragma unroll
  for (int mt = 0; mt < 4; ++mt)
    #pragma unroll
    for (int r = 0; r < 8; ++r)
      A[(mt * 16 + r + ((lane >> 4) << 3)) * 16 + col] = (_Float16)ct[mt][r];
  __syncthreads();

  // GEMM4: out = Wp @ t + b_proj
  #pragma unroll
  for (int mt = 0; mt < 4; ++mt) {
    v8f c = {};
    #pragma unroll
    for (int k0 = 0; k0 < 64; k0 += 32) {
      v16h a  = frag_a(&Wp[0][0], 64, mt * 16, k0, lane);
      v16h bb = frag_b(A, 16, k0, 0, lane);
      c = __builtin_amdgcn_wmma_f32_16x16x32_f16(false, a, false, bb,
                                                 (short)0, c, false, false);
    }
    #pragma unroll
    for (int r = 0; r < 8; ++r) {
      const int row = mt * 16 + r + ((lane >> 4) << 3);
      out[((size_t)b * CH + row) * NTOK + p0 + col] = c[r] + b_proj[row];
    }
  }
}

// ---------------------------------------------------------------------------
extern "C" void kernel_launch(void* const* d_in, const int* in_sizes, int n_in,
                              void* d_out, int out_size, void* d_ws, size_t ws_size,
                              hipStream_t stream) {
  (void)in_sizes; (void)n_in; (void)out_size; (void)ws_size;
  const float* x           = (const float*)d_in[0];
  const float* w_qkv       = (const float*)d_in[1];
  const float* b_qkv       = (const float*)d_in[2];
  const float* w_dw        = (const float*)d_in[3];
  const float* b_dw        = (const float*)d_in[4];
  const float* temperature = (const float*)d_in[5];
  const float* attn_w      = (const float*)d_in[6];
  const float* w_recal     = (const float*)d_in[7];
  const float* b_recal     = (const float*)d_in[8];
  const float* bn1_g       = (const float*)d_in[9];
  const float* bn1_b       = (const float*)d_in[10];
  const float* w_fus       = (const float*)d_in[11];
  const float* b_fus       = (const float*)d_in[12];
  const float* bn2_g       = (const float*)d_in[13];
  const float* bn2_b       = (const float*)d_in[14];
  const float* w_proj      = (const float*)d_in[15];
  const float* b_proj      = (const float*)d_in[16];
  float* out = (float*)d_out;

  char* ws = (char*)d_ws;
  const size_t qkvh_bytes  = (size_t)BATCH * QKV_CH * NTOK * sizeof(_Float16); // 50.3 MB
  const size_t qkvdw_bytes = (size_t)BATCH * QKV_CH * NTOK * sizeof(float);    // 100.7 MB
  _Float16* qkv_h = (_Float16*)ws;
  float* qkvdw    = (float*)(ws + qkvh_bytes);
  float* stats    = (float*)(ws + qkvh_bytes + qkvdw_bytes);
  float* P        = stats + BATCH * HEADS * 80;

  hipMemsetAsync(stats, 0, (size_t)BATCH * HEADS * 80 * sizeof(float), stream);

  k_qkv<<<BATCH * (NTOK / 64), 256, 0, stream>>>(x, w_qkv, b_qkv, qkv_h);

  const long total = (long)BATCH * QKV_CH * NTOK;
  const int dw_blocks = (int)((total + 255) / 256);
  k_dwconv<<<dw_blocks, 256, 0, stream>>>(qkv_h, w_dw, b_dw, qkvdw);

  k_gram<<<BATCH * HEADS * 64, 64, 0, stream>>>(qkvdw, stats);
  k_combine<<<1, 128, 0, stream>>>(stats, temperature, attn_w, P);

  k_tail<<<BATCH * (NTOK / 64), 128, 0, stream>>>(qkvdw, x, P,
      w_recal, b_recal, bn1_g, bn1_b, w_fus, b_fus, bn2_g, bn2_b,
      w_proj, b_proj, out);
}
